// HGTIntra_37864431682358
// MI455X (gfx1250) — compile-verified, matrix operand order checked by reading, not run on hardware
//
#include <hip/hip_runtime.h>
#include <hip/hip_bf16.h>
#include <stdint.h>

// ---------------------------------------------------------------------------
// HGT forward for MI455X (gfx1250, wave32, WMMA + TDM).
// - Dense projections: 64x64-tile WMMA bf16 GEMMs, f32 accumulate.
// - Activations kept with persistent bf16 mirrors (halves GEMM A-traffic).
// - K/V projections fused (one A pass, 4 WMMAs/iter), rel matrices folded
//   into the weights on device (bf16).
// - A tiles staged into LDS with the Tensor Data Mover (tensor_load_to_lds,
//   s_wait_tensorcnt), padded to the [BK+2] LDS row stride via D# padding.
// - Edge attention: 3-pass segmented softmax with atomics (fits in 192MB L2).
// ---------------------------------------------------------------------------

typedef __attribute__((ext_vector_type(16))) __bf16 bfrag_t;  // A/B fragment
typedef __attribute__((ext_vector_type(8)))  float  v8f;      // C/D fragment
typedef __attribute__((ext_vector_type(4)))  unsigned int v4u;
typedef __attribute__((ext_vector_type(8)))  int v8i;
typedef __attribute__((ext_vector_type(4)))  int v4i;

#define BM 64
#define BN 64
#define BK 32

struct Frag { unsigned int u[8]; };
union U16x8 { uint4 q; unsigned short s[8]; };

__device__ __forceinline__ unsigned short f32_to_bf16(float f) {
  unsigned int u = __float_as_uint(f);
  unsigned int r = u + 0x7FFFu + ((u >> 16) & 1u);  // round-to-nearest-even
  return (unsigned short)(r >> 16);
}

__device__ __forceinline__ float gelu_exact(float x) {
  return 0.5f * x * (1.0f + erff(x * 0.70710678118654752f));
}

// Ordered-uint encoding so atomicMax(uint) implements float max (init = 0).
__device__ __forceinline__ unsigned f32_ord(float f) {
  unsigned u = __float_as_uint(f);
  return (u & 0x80000000u) ? ~u : (u | 0x80000000u);
}
__device__ __forceinline__ float ord_f32(unsigned e) {
  return (e & 0x80000000u) ? __uint_as_float(e ^ 0x80000000u)
                           : __uint_as_float(~e);
}

// ---------------------------------------------------------------------------
// TDM: async-load a [rows<=64] x 32 bf16 tile (row stride = stride_elems)
// into LDS at lds_addr, with 1 dword of padding after each 16-dword (64B) row
// so the LDS row stride matches As[BM][BK+2]. OOB rows return zeros.
// D# layout per CDNA5 ISA 8.3/8.4 (group0: count/lds/global/type,
// group1: data_size=2B, pad_en, pad_interval=16dw, pad_amount=1dw, dims).
// ---------------------------------------------------------------------------
__device__ __forceinline__ void tdm_load_tile_bf16(
    unsigned int lds_addr, const unsigned short* gptr, unsigned int rows,
    unsigned int stride_elems) {
  unsigned long long ga = (unsigned long long)(size_t)gptr;
  v4u g0;
  g0[0] = 1u;  // count=1 (valid user descriptor)
  g0[1] = lds_addr;
  g0[2] = (unsigned int)(ga & 0xFFFFFFFFull);
  g0[3] = (unsigned int)((ga >> 32) & 0x1FFFFFFull) | (2u << 30);  // type=2
  v8i g1;
  g1[0] = (int)((1u << 16) |   // data_size: 2 bytes
                (1u << 20) |   // pad_enable
                (3u << 22) |   // pad_interval: 16 dwords
                (0u << 25));   // pad_amount: 1 dword
  g1[1] = (int)(32u << 16);                              // tensor_dim0 lo16
  g1[2] = (int)((rows & 0xFFFFu) << 16);                 // tensor_dim1 lo16
  g1[3] = (int)(((rows >> 16) & 0xFFFFu) | (32u << 16)); // dim1 hi | tile_dim0
  g1[4] = 64;                                            // tile_dim1 (tile_dim2=0)
  g1[5] = (int)stride_elems;                             // tensor_dim0_stride lo32
  g1[6] = 0;
  g1[7] = 0;
  v4i z = {0, 0, 0, 0};
#if __clang_major__ >= 23
  v8i z8 = {0, 0, 0, 0, 0, 0, 0, 0};
  __builtin_amdgcn_tensor_load_to_lds(g0, g1, z, z, z8, 0);
#else
  __builtin_amdgcn_tensor_load_to_lds(g0, g1, z, z, 0);
#endif
}

// Build one WMMA A/B fragment from an LDS tile with [BK+2] row stride,
// per the CDNA5 16-bit fragment layout (K pairs; lanes>=16 get K+8).
__device__ __forceinline__ bfrag_t load_frag(
    const unsigned short (*T)[BK + 2], int rowsel, int koff) {
  Frag f;
#pragma unroll
  for (int j = 0; j < 8; ++j) {
    int kk = ((j < 4) ? (j << 1) : (16 + ((j - 4) << 1))) + koff;
    f.u[j] = *(const unsigned int*)&T[rowsel][kk];
  }
  return __builtin_bit_cast(bfrag_t, f);
}

// ---------------------------------------------------------------------------
// GEMM (fp32 A): C = act(A@W + bias); optional bf16 mirror of C.
// Used for the input adaptation (f0/f1 are fp32).
// ---------------------------------------------------------------------------
template <int ACT>
__global__ __launch_bounds__(256) void gemm_af32_kernel(
    const float* __restrict__ A, const float* __restrict__ W,
    const float* __restrict__ bias, float* __restrict__ C,
    unsigned short* __restrict__ Cbf, int M, int K, int Nc) {
  __shared__ unsigned short As[BM][BK + 2];
  __shared__ unsigned short Bs[BN][BK + 2];
  const int tid = threadIdx.x;
  const int lane = tid & 31;
  const int wave = tid >> 5;
  const int wr = wave >> 1, wc = wave & 1;
  const int m0 = blockIdx.x * BM, n0 = blockIdx.y * BN;
  const int halfsel = (lane >> 4) & 1;
  const int koff = halfsel * 8;
  const int l15 = lane & 15;
  v8f acc0 = {}, acc1 = {};

  for (int k0 = 0; k0 < K; k0 += BK) {
#pragma unroll
    for (int it = 0; it < 2; ++it) {     // A tile 64x32 f32 -> bf16
      int idx = tid + it * 256;
      int r = idx >> 3, c4 = (idx & 7) << 2;
      float4 v = make_float4(0.f, 0.f, 0.f, 0.f);
      int gr = m0 + r;
      if (gr < M) v = *(const float4*)(A + (size_t)gr * K + k0 + c4);
      As[r][c4 + 0] = f32_to_bf16(v.x);
      As[r][c4 + 1] = f32_to_bf16(v.y);
      As[r][c4 + 2] = f32_to_bf16(v.z);
      As[r][c4 + 3] = f32_to_bf16(v.w);
    }
#pragma unroll
    for (int it = 0; it < 2; ++it) {     // B tile 32x64, store transposed
      int idx = tid + it * 256;
      int r = idx >> 4, c4 = (idx & 15) << 2;
      float4 v = *(const float4*)(W + (size_t)(k0 + r) * Nc + n0 + c4);
      Bs[c4 + 0][r] = f32_to_bf16(v.x);
      Bs[c4 + 1][r] = f32_to_bf16(v.y);
      Bs[c4 + 2][r] = f32_to_bf16(v.z);
      Bs[c4 + 3][r] = f32_to_bf16(v.w);
    }
    __syncthreads();
    bfrag_t a  = load_frag(As, (wr << 4) + l15, koff);
    bfrag_t b0 = load_frag(Bs, (wc << 5) + l15, koff);
    bfrag_t b1 = load_frag(Bs, (wc << 5) + 16 + l15, koff);
    acc0 = __builtin_amdgcn_wmma_f32_16x16x32_bf16(false, a, false, b0,
                                                   (short)0, acc0, false, false);
    acc1 = __builtin_amdgcn_wmma_f32_16x16x32_bf16(false, a, false, b1,
                                                   (short)0, acc1, false, false);
    __syncthreads();
  }

  const int rowbase = m0 + (wr << 4) + halfsel * 8;
  const int col0 = n0 + (wc << 5) + l15;
  const int col1 = col0 + 16;
  const float bi0 = bias[col0], bi1 = bias[col1];
#pragma unroll
  for (int r = 0; r < 8; ++r) {
    int row = rowbase + r;
    if (row < M) {
      float x0 = acc0[r] + bi0;
      float x1 = acc1[r] + bi1;
      if (ACT == 1) { x0 = gelu_exact(x0); x1 = gelu_exact(x1); }
      C[(size_t)row * Nc + col0] = x0;
      C[(size_t)row * Nc + col1] = x1;
      if (Cbf) {
        Cbf[(size_t)row * Nc + col0] = f32_to_bf16(x0);
        Cbf[(size_t)row * Nc + col1] = f32_to_bf16(x1);
      }
    }
  }
}

// ---------------------------------------------------------------------------
// GEMM (bf16 A via TDM): C = A@W + bias, W fp32 converted while staging.
// Used for Q, a_w (A = agg bf16) and output-embedding projections.
// ---------------------------------------------------------------------------
__global__ __launch_bounds__(256) void gemm_abf16_kernel(
    const unsigned short* __restrict__ Abf, const float* __restrict__ W,
    const float* __restrict__ bias, float* __restrict__ C, int M, int K,
    int Nc) {
  __shared__ unsigned short As[BM][BK + 2];
  __shared__ unsigned short Bs[BN][BK + 2];
  const int tid = threadIdx.x;
  const int lane = tid & 31;
  const int wave = tid >> 5;
  const int wr = wave >> 1, wc = wave & 1;
  const int m0 = blockIdx.x * BM, n0 = blockIdx.y * BN;
  const int halfsel = (lane >> 4) & 1;
  const int koff = halfsel * 8;
  const int l15 = lane & 15;
  v8f acc0 = {}, acc1 = {};
  const unsigned int ldsA = (unsigned int)(size_t)&As[0][0];
  const unsigned int rows = (unsigned int)(M - m0);

  for (int k0 = 0; k0 < K; k0 += BK) {
    if (wave == 0)
      tdm_load_tile_bf16(ldsA, Abf + (size_t)m0 * K + k0, rows, (unsigned)K);
#pragma unroll
    for (int it = 0; it < 2; ++it) {     // B tile 32x64 f32 -> bf16 transposed
      int idx = tid + it * 256;
      int r = idx >> 4, c4 = (idx & 15) << 2;
      float4 v = *(const float4*)(W + (size_t)(k0 + r) * Nc + n0 + c4);
      Bs[c4 + 0][r] = f32_to_bf16(v.x);
      Bs[c4 + 1][r] = f32_to_bf16(v.y);
      Bs[c4 + 2][r] = f32_to_bf16(v.z);
      Bs[c4 + 3][r] = f32_to_bf16(v.w);
    }
    if (wave == 0) __builtin_amdgcn_s_wait_tensorcnt(0);
    __syncthreads();
    bfrag_t a  = load_frag(As, (wr << 4) + l15, koff);
    bfrag_t b0 = load_frag(Bs, (wc << 5) + l15, koff);
    bfrag_t b1 = load_frag(Bs, (wc << 5) + 16 + l15, koff);
    acc0 = __builtin_amdgcn_wmma_f32_16x16x32_bf16(false, a, false, b0,
                                                   (short)0, acc0, false, false);
    acc1 = __builtin_amdgcn_wmma_f32_16x16x32_bf16(false, a, false, b1,
                                                   (short)0, acc1, false, false);
    __syncthreads();
  }

  const int rowbase = m0 + (wr << 4) + halfsel * 8;
  const int col0 = n0 + (wc << 5) + l15;
  const int col1 = col0 + 16;
  const float bi0 = bias[col0], bi1 = bias[col1];
#pragma unroll
  for (int r = 0; r < 8; ++r) {
    int row = rowbase + r;
    if (row < M) {
      C[(size_t)row * Nc + col0] = acc0[r] + bi0;
      C[(size_t)row * Nc + col1] = acc1[r] + bi1;
    }
  }
}

// ---------------------------------------------------------------------------
// Fused K+V GEMM (bf16 A via TDM, bf16 weights): one A pass, 4 WMMAs/iter.
// Nc fixed at 256.
// ---------------------------------------------------------------------------
__global__ __launch_bounds__(256) void gemm_kv_kernel(
    const unsigned short* __restrict__ Abf,
    const unsigned short* __restrict__ Wk,
    const unsigned short* __restrict__ Wv, const float* __restrict__ bk,
    const float* __restrict__ bv, float* __restrict__ Ck,
    float* __restrict__ Cv, int M, int K) {
  __shared__ unsigned short As[BM][BK + 2];
  __shared__ unsigned short Bks[BN][BK + 2];
  __shared__ unsigned short Bvs[BN][BK + 2];
  const int tid = threadIdx.x;
  const int lane = tid & 31;
  const int wave = tid >> 5;
  const int wr = wave >> 1, wc = wave & 1;
  const int m0 = blockIdx.x * BM, n0 = blockIdx.y * BN;
  const int halfsel = (lane >> 4) & 1;
  const int koff = halfsel * 8;
  const int l15 = lane & 15;
  v8f ak0 = {}, ak1 = {}, av0 = {}, av1 = {};
  const unsigned int ldsA = (unsigned int)(size_t)&As[0][0];
  const unsigned int rows = (unsigned int)(M - m0);

  const int br = tid >> 3;            // k row 0..31
  const int bc = (tid & 7) << 3;      // n 0..56 (8 bf16 per thread)

  for (int k0 = 0; k0 < K; k0 += BK) {
    if (wave == 0)
      tdm_load_tile_bf16(ldsA, Abf + (size_t)m0 * K + k0, rows, (unsigned)K);
    {   // stage both weight tiles (already bf16): pure transposed copies
      U16x8 uk, uv;
      uk.q = *(const uint4*)(Wk + (size_t)(k0 + br) * 256 + n0 + bc);
      uv.q = *(const uint4*)(Wv + (size_t)(k0 + br) * 256 + n0 + bc);
#pragma unroll
      for (int j = 0; j < 8; ++j) {
        Bks[bc + j][br] = uk.s[j];
        Bvs[bc + j][br] = uv.s[j];
      }
    }
    if (wave == 0) __builtin_amdgcn_s_wait_tensorcnt(0);
    __syncthreads();
    bfrag_t a   = load_frag(As, (wr << 4) + l15, koff);
    bfrag_t bk0 = load_frag(Bks, (wc << 5) + l15, koff);
    bfrag_t bk1 = load_frag(Bks, (wc << 5) + 16 + l15, koff);
    bfrag_t bv0 = load_frag(Bvs, (wc << 5) + l15, koff);
    bfrag_t bv1 = load_frag(Bvs, (wc << 5) + 16 + l15, koff);
    ak0 = __builtin_amdgcn_wmma_f32_16x16x32_bf16(false, a, false, bk0,
                                                  (short)0, ak0, false, false);
    ak1 = __builtin_amdgcn_wmma_f32_16x16x32_bf16(false, a, false, bk1,
                                                  (short)0, ak1, false, false);
    av0 = __builtin_amdgcn_wmma_f32_16x16x32_bf16(false, a, false, bv0,
                                                  (short)0, av0, false, false);
    av1 = __builtin_amdgcn_wmma_f32_16x16x32_bf16(false, a, false, bv1,
                                                  (short)0, av1, false, false);
    __syncthreads();
  }

  const int rowbase = m0 + (wr << 4) + halfsel * 8;
  const int col0 = n0 + (wc << 5) + l15;
  const int col1 = col0 + 16;
  const float bk0f = bk[col0], bk1f = bk[col1];
  const float bv0f = bv[col0], bv1f = bv[col1];
#pragma unroll
  for (int r = 0; r < 8; ++r) {
    int row = rowbase + r;
    if (row < M) {
      Ck[(size_t)row * 256 + col0] = ak0[r] + bk0f;
      Ck[(size_t)row * 256 + col1] = ak1[r] + bk1f;
      Cv[(size_t)row * 256 + col0] = av0[r] + bv0f;
      Cv[(size_t)row * 256 + col1] = av1[r] + bv1f;
    }
  }
}

// ---------------------------------------------------------------------------
// Fold per-head rel matrix into projection weight (bf16 out):
//   wf[i][h*32+e] = sum_d w[i][h*32+d] * rel[h][d][e]   (row 256 does bias)
// ---------------------------------------------------------------------------
__global__ __launch_bounds__(256) void fuse_weight_kernel(
    const float* __restrict__ w, const float* __restrict__ bsrc,
    const float* __restrict__ rel, unsigned short* __restrict__ wf,
    float* __restrict__ bf) {
  int i = blockIdx.x;        // 0..255 weight rows, 256 == bias row
  int j = threadIdx.x;       // output column
  int h = j >> 5, e = j & 31;
  const float* row = (i < 256) ? (w + (size_t)i * 256) : bsrc;
  float acc = 0.f;
#pragma unroll
  for (int d = 0; d < 32; ++d)
    acc += row[h * 32 + d] * rel[h * 1024 + d * 32 + e];
  if (i < 256) wf[(size_t)i * 256 + j] = f32_to_bf16(acc);
  else         bf[j] = acc;
}

// ---------------------------------------------------------------------------
// Edge score: one wave32 per edge; lane covers 8 contiguous feature elems
// (all within head lane/4), reduce across 4 lanes per head.
// ---------------------------------------------------------------------------
__global__ __launch_bounds__(256) void edge_score_kernel(
    const float* __restrict__ q, const float* __restrict__ k,
    const int* __restrict__ src, const int* __restrict__ dst,
    const float* __restrict__ rel_pri, float* __restrict__ score, int E) {
  int e = blockIdx.x * 8 + (threadIdx.x >> 5);
  int lane = threadIdx.x & 31;
  if (e >= E) return;
  int s = src[e], d = dst[e];
  const float4* qp = (const float4*)(q + (size_t)d * 256 + lane * 8);
  const float4* kp = (const float4*)(k + (size_t)s * 256 + lane * 8);
  float4 q0 = qp[0], q1 = qp[1];
  float4 k0 = kp[0], k1 = kp[1];
  float p = q0.x * k0.x + q0.y * k0.y + q0.z * k0.z + q0.w * k0.w +
            q1.x * k1.x + q1.y * k1.y + q1.z * k1.z + q1.w * k1.w;
  p += __shfl_xor(p, 1);
  p += __shfl_xor(p, 2);
  if ((lane & 3) == 0) {
    int h = lane >> 2;
    score[(size_t)e * 8 + h] = p * rel_pri[h] * 0.17677669529663687f; // 1/sqrt(32)
  }
}

__global__ __launch_bounds__(256) void edge_max_kernel(
    const float* __restrict__ score, const int* __restrict__ dst,
    unsigned* __restrict__ mx, int EH) {
  int i = blockIdx.x * 256 + threadIdx.x;
  if (i >= EH) return;
  int e = i >> 3, h = i & 7;
  atomicMax(&mx[(size_t)dst[e] * 8 + h], f32_ord(score[i]));
}

__global__ __launch_bounds__(256) void edge_exp_kernel(
    float* __restrict__ score, const int* __restrict__ dst,
    const unsigned* __restrict__ mx, float* __restrict__ den, int EH) {
  int i = blockIdx.x * 256 + threadIdx.x;
  if (i >= EH) return;
  int e = i >> 3, h = i & 7;
  int d = dst[e];
  float m = ord_f32(mx[(size_t)d * 8 + h]);
  float ex = __expf(score[i] - m);
  score[i] = ex;
  atomicAdd(&den[(size_t)d * 8 + h], ex);
}

// Unnormalized message scatter: agg[dst] += v[src] * exp(score-max)
__global__ __launch_bounds__(256) void edge_msg_kernel(
    const float* __restrict__ v, const float* __restrict__ ex,
    const int* __restrict__ src, const int* __restrict__ dst,
    float* __restrict__ agg, int E) {
  int e = blockIdx.x * 8 + (threadIdx.x >> 5);
  int lane = threadIdx.x & 31;
  if (e >= E) return;
  int s = src[e], d = dst[e];
  float a = ex[(size_t)e * 8 + (lane >> 2)];
  const float4* vp = (const float4*)(v + (size_t)s * 256 + lane * 8);
  float4 v0 = vp[0], v1 = vp[1];
  float* ap = agg + (size_t)d * 256 + lane * 8;
  atomicAdd(ap + 0, v0.x * a); atomicAdd(ap + 1, v0.y * a);
  atomicAdd(ap + 2, v0.z * a); atomicAdd(ap + 3, v0.w * a);
  atomicAdd(ap + 4, v1.x * a); atomicAdd(ap + 5, v1.y * a);
  atomicAdd(ap + 6, v1.z * a); atomicAdd(ap + 7, v1.w * a);
}

// Normalize by softmax denominator and emit bf16 (feeds a_w WMMA GEMM).
__global__ __launch_bounds__(256) void agg_norm_kernel(
    const float* __restrict__ agg, const float* __restrict__ den,
    unsigned short* __restrict__ out, int total) {
  int i = blockIdx.x * 256 + threadIdx.x;
  if (i >= total) return;
  int n = i >> 8, c = i & 255, h = c >> 5;
  float dn = den[(size_t)n * 8 + h];
  if (dn == 0.f) dn = 1.f;
  out[i] = f32_to_bf16(agg[i] / dn);
}

// ---------------------------------------------------------------------------
// Skip + LayerNorm: one wave per row of 256; updates h (f32) in place and
// writes the bf16 mirror used as WMMA A in the next layer.
// ---------------------------------------------------------------------------
__global__ __launch_bounds__(256) void ln_skip_kernel(
    const float* __restrict__ trans, float* __restrict__ h,
    unsigned short* __restrict__ hbf, const float* __restrict__ skipv,
    const float* __restrict__ g, const float* __restrict__ b, int Nn) {
  int n = blockIdx.x * 8 + (threadIdx.x >> 5);
  int lane = threadIdx.x & 31;
  if (n >= Nn) return;
  float alpha = 1.f / (1.f + __expf(-skipv[0]));
  float ia = 1.f - alpha;
  const float4* tp = (const float4*)(trans + (size_t)n * 256 + lane * 8);
  float4* hp = (float4*)(h + (size_t)n * 256 + lane * 8);
  float4 t0 = tp[0], t1 = tp[1];
  float4 h0 = hp[0], h1 = hp[1];
  float o[8];
  o[0] = t0.x * alpha + h0.x * ia; o[1] = t0.y * alpha + h0.y * ia;
  o[2] = t0.z * alpha + h0.z * ia; o[3] = t0.w * alpha + h0.w * ia;
  o[4] = t1.x * alpha + h1.x * ia; o[5] = t1.y * alpha + h1.y * ia;
  o[6] = t1.z * alpha + h1.z * ia; o[7] = t1.w * alpha + h1.w * ia;
  float sm = 0.f, sq = 0.f;
#pragma unroll
  for (int j = 0; j < 8; ++j) { sm += o[j]; sq += o[j] * o[j]; }
#pragma unroll
  for (int off = 1; off < 32; off <<= 1) {
    sm += __shfl_xor(sm, off);
    sq += __shfl_xor(sq, off);
  }
  float mu = sm * (1.f / 256.f);
  float var = sq * (1.f / 256.f) - mu * mu;
  float rs = rsqrtf(var + 1e-5f);
  int cb = lane * 8;
  float r[8];
#pragma unroll
  for (int j = 0; j < 8; ++j) r[j] = (o[j] - mu) * rs * g[cb + j] + b[cb + j];
  hp[0] = make_float4(r[0], r[1], r[2], r[3]);
  hp[1] = make_float4(r[4], r[5], r[6], r[7]);
  U16x8 pk;
#pragma unroll
  for (int j = 0; j < 8; ++j) pk.s[j] = f32_to_bf16(r[j]);
  *(uint4*)(hbf + (size_t)n * 256 + cb) = pk.q;
}

// ---------------------------------------------------------------------------
// Pooling: segment-sum emb (+masked variants) into per-graph vectors.
// ---------------------------------------------------------------------------
__global__ __launch_bounds__(256) void pool_kernel(
    const float* __restrict__ emb, const int* __restrict__ m1,
    const int* __restrict__ m2, float* __restrict__ pall,
    float* __restrict__ p1, float* __restrict__ p2, int Nn, int npg) {
  int i = blockIdx.x * 256 + threadIdx.x;
  if (i >= Nn * 64) return;
  int n = i >> 6, c = i & 63;
  int g = n / npg;
  float v = emb[i];
  atomicAdd(&pall[g * 64 + c], v);
  if (m1[n]) atomicAdd(&p1[g * 64 + c], v);
  if (m2[n]) atomicAdd(&p2[g * 64 + c], v);
}

// Final heads: out[g][c] = bias[c] + concat(p0[g],p1[g]) . W[:,c]
__global__ __launch_bounds__(64) void head_kernel(
    const float* __restrict__ p0, const float* __restrict__ p1,
    const float* __restrict__ W, const float* __restrict__ b,
    float* __restrict__ out) {
  int g = blockIdx.x, c = threadIdx.x;
  float acc = b[c];
#pragma unroll 4
  for (int i = 0; i < 64; ++i) acc += p0[g * 64 + i] * W[i * 64 + c];
#pragma unroll 4
  for (int i = 0; i < 64; ++i) acc += p1[g * 64 + i] * W[(64 + i) * 64 + c];
  out[g * 64 + c] = acc;
}

__global__ __launch_bounds__(256) void zero_kernel(float* __restrict__ p,
                                                   size_t n) {
  size_t i = (size_t)blockIdx.x * 256 + threadIdx.x;
  if (i < n) p[i] = 0.f;
}

// ---------------------------------------------------------------------------
// Orchestration. Input order = JAX pytree (sorted-key) flatten of the inputs:
//  0:f0 1:f1 2:src0 3:dst0 4:src1 5:dst1 6:r1m0 7:r1m1 8:r2m0 9:r2m1
//  10:adapt_b 11:adapt_w
//  layers: 12:a_b 13:a_w 14:k_b 15:k_w 16:ln_b 17:ln_g 18:q_b 19:q_w
//          20:rel_att 21:rel_msg 22:rel_pri 23:skip 24:v_b 25:v_w
//  26:one_b 27:one_w 28:out_b 29:out_w 30:pn_b 31:pn_w 32:reg_b 33:reg_w
// ---------------------------------------------------------------------------
extern "C" void kernel_launch(void* const* d_in, const int* in_sizes, int n_in,
                              void* d_out, int out_size, void* d_ws,
                              size_t ws_size, hipStream_t stream) {
  (void)n_in; (void)out_size; (void)ws_size;
  const float* f0 = (const float*)d_in[0];
  const float* f1 = (const float*)d_in[1];
  const int* srcR[2] = {(const int*)d_in[2], (const int*)d_in[4]};
  const int* dstR[2] = {(const int*)d_in[3], (const int*)d_in[5]};
  const int* r1m[2] = {(const int*)d_in[6], (const int*)d_in[7]};
  const int* r2m[2] = {(const int*)d_in[8], (const int*)d_in[9]};
  const float* adapt_b = (const float*)d_in[10];
  const float* adapt_w = (const float*)d_in[11];
  const float* a_b  = (const float*)d_in[12];
  const float* a_w  = (const float*)d_in[13];
  const float* k_b  = (const float*)d_in[14];
  const float* k_w  = (const float*)d_in[15];
  const float* ln_b = (const float*)d_in[16];
  const float* ln_g = (const float*)d_in[17];
  const float* q_b  = (const float*)d_in[18];
  const float* q_w  = (const float*)d_in[19];
  const float* rel_att = (const float*)d_in[20];
  const float* rel_msg = (const float*)d_in[21];
  const float* rel_pri = (const float*)d_in[22];
  const float* skip = (const float*)d_in[23];
  const float* v_b  = (const float*)d_in[24];
  const float* v_w  = (const float*)d_in[25];
  const float* one_b = (const float*)d_in[26];
  const float* one_w = (const float*)d_in[27];
  const float* out_b = (const float*)d_in[28];
  const float* out_w = (const float*)d_in[29];
  const float* pn_b  = (const float*)d_in[30];
  const float* pn_w  = (const float*)d_in[31];
  const float* reg_b = (const float*)d_in[32];
  const float* reg_w = (const float*)d_in[33];

  const int N = in_sizes[0] / 256;   // 50000
  const int E = in_sizes[2];         // 400000
  const int npg = N / 8;             // nodes per graph

  // ---- workspace carve (256B aligned)
  char* wp = (char*)d_ws;
  auto alloc = [&](size_t bytes) -> char* {
    char* r = wp;
    wp += (bytes + 255) & ~(size_t)255;
    return r;
  };
  const size_t HS  = (size_t)N * 256 * sizeof(float);
  const size_t HSB = (size_t)N * 256 * sizeof(unsigned short);
  float* h[2]   = {(float*)alloc(HS), (float*)alloc(HS)};
  unsigned short* hbf[2] = {(unsigned short*)alloc(HSB),
                            (unsigned short*)alloc(HSB)};
  float* kbuf   = (float*)alloc(HS);   // also reused as tmp / emb0
  float* qbuf   = (float*)alloc(HS);   // also reused as emb1
  float* vbuf   = (float*)alloc(HS);
  float* agg[2] = {(float*)alloc(HS), (float*)alloc(HS)};
  unsigned short* aggbf[2] = {(unsigned short*)alloc(HSB),
                              (unsigned short*)alloc(HSB)};
  float* ebuf   = (float*)alloc((size_t)E * 8 * sizeof(float));
  float* den    = (float*)alloc((size_t)N * 8 * sizeof(float));
  unsigned* mxb = (unsigned*)alloc((size_t)N * 8 * sizeof(unsigned));
  unsigned short* wfk = (unsigned short*)alloc(65536 * sizeof(unsigned short));
  float* bfk = (float*)alloc(256 * sizeof(float));
  unsigned short* wfv = (unsigned short*)alloc(65536 * sizeof(unsigned short));
  float* bfv = (float*)alloc(256 * sizeof(float));
  float* pools = (float*)alloc(6 * 8 * 64 * sizeof(float));

  const dim3 blk(256);
  const int gm = (N + BM - 1) / BM;
  const int EH = E * 8;
  const int ehBlocks = (EH + 255) / 256;
  const int edgeBlocks = (E + 7) / 8;
  const int nhBlocks = (N * 256 + 255) / 256;

  // ---- input adaptation: h[t] = gelu(f @ adapt_w[t] + adapt_b[t]) + bf16 mirror
  gemm_af32_kernel<1><<<dim3(gm, 4), blk, 0, stream>>>(
      f0, adapt_w, adapt_b, h[0], hbf[0], N, 256, 256);
  gemm_af32_kernel<1><<<dim3(gm, 4), blk, 0, stream>>>(
      f1, adapt_w + 65536, adapt_b + 256, h[1], hbf[1], N, 256, 256);

  const int rel_s[2] = {0, 1}, rel_d[2] = {1, 0};
  for (int l = 0; l < 2; ++l) {
    for (int r = 0; r < 2; ++r) {
      const int s = rel_s[r], d = rel_d[r];
      const float* kw = k_w + (size_t)(l * 2 + s) * 65536;
      const float* kb = k_b + (size_t)(l * 2 + s) * 256;
      const float* vw = v_w + (size_t)(l * 2 + s) * 65536;
      const float* vb = v_b + (size_t)(l * 2 + s) * 256;
      const float* qw = q_w + (size_t)(l * 2 + d) * 65536;
      const float* qb = q_b + (size_t)(l * 2 + d) * 256;
      const float* ra = rel_att + (size_t)(l * 2 + r) * 8192;
      const float* rm = rel_msg + (size_t)(l * 2 + r) * 8192;
      const float* rp = rel_pri + (size_t)(l * 2 + r) * 8;

      fuse_weight_kernel<<<257, blk, 0, stream>>>(kw, kb, ra, wfk, bfk);
      fuse_weight_kernel<<<257, blk, 0, stream>>>(vw, vb, rm, wfv, bfv);

      gemm_kv_kernel<<<dim3(gm, 4), blk, 0, stream>>>(
          hbf[s], wfk, wfv, bfk, bfv, kbuf, vbuf, N, 256);
      gemm_abf16_kernel<<<dim3(gm, 4), blk, 0, stream>>>(
          hbf[d], qw, qb, qbuf, N, 256, 256);

      zero_kernel<<<(N * 8 + 255) / 256, blk, 0, stream>>>((float*)mxb,
                                                           (size_t)N * 8);
      zero_kernel<<<(N * 8 + 255) / 256, blk, 0, stream>>>(den, (size_t)N * 8);
      zero_kernel<<<nhBlocks, blk, 0, stream>>>(agg[d], (size_t)N * 256);

      edge_score_kernel<<<edgeBlocks, blk, 0, stream>>>(
          qbuf, kbuf, srcR[r], dstR[r], rp, ebuf, E);
      edge_max_kernel<<<ehBlocks, blk, 0, stream>>>(ebuf, dstR[r], mxb, EH);
      edge_exp_kernel<<<ehBlocks, blk, 0, stream>>>(ebuf, dstR[r], mxb, den, EH);
      edge_msg_kernel<<<edgeBlocks, blk, 0, stream>>>(
          vbuf, ebuf, srcR[r], dstR[r], agg[d], E);
      agg_norm_kernel<<<nhBlocks, blk, 0, stream>>>(agg[d], den, aggbf[d],
                                                    N * 256);
    }
    for (int t = 0; t < 2; ++t) {
      gemm_abf16_kernel<<<dim3(gm, 4), blk, 0, stream>>>(
          aggbf[t], a_w + (size_t)(l * 2 + t) * 65536,
          a_b + (size_t)(l * 2 + t) * 256, kbuf, N, 256, 256);
      ln_skip_kernel<<<(N + 7) / 8, blk, 0, stream>>>(
          kbuf, h[t], hbf[t], skip + (l * 2 + t),
          ln_g + (size_t)(l * 2 + t) * 256,
          ln_b + (size_t)(l * 2 + t) * 256, N);
    }
  }

  // ---- output embeddings [N,64] per ntype (reuse kbuf/qbuf)
  gemm_abf16_kernel<<<dim3(gm, 1), blk, 0, stream>>>(
      hbf[0], out_w, out_b, kbuf, N, 256, 64);
  gemm_abf16_kernel<<<dim3(gm, 1), blk, 0, stream>>>(
      hbf[1], out_w + 16384, out_b + 64, qbuf, N, 256, 64);

  zero_kernel<<<(6 * 512 + 255) / 256, blk, 0, stream>>>(pools, 6 * 512);
  const int pBlocks = (N * 64 + 255) / 256;
  pool_kernel<<<pBlocks, blk, 0, stream>>>(
      kbuf, r1m[0], r2m[0], pools + 0 * 512, pools + 2 * 512, pools + 4 * 512,
      N, npg);
  pool_kernel<<<pBlocks, blk, 0, stream>>>(
      qbuf, r1m[1], r2m[1], pools + 1 * 512, pools + 3 * 512, pools + 5 * 512,
      N, npg);

  float* out = (float*)d_out;
  head_kernel<<<8, 64, 0, stream>>>(pools + 0 * 512, pools + 1 * 512, one_w,
                                    one_b, out + 0);
  head_kernel<<<8, 64, 0, stream>>>(pools + 2 * 512, pools + 3 * 512, reg_w,
                                    reg_b, out + 512);
  head_kernel<<<8, 64, 0, stream>>>(pools + 4 * 512, pools + 5 * 512, pn_w,
                                    pn_b, out + 1024);
}
